// BasicBlock_refine_27307402068756
// MI455X (gfx1250) — matrix operand order repaired and checked by reading, not compiled, not run on hardware
//
#include <hip/hip_runtime.h>

// ---- CDNA5 WMMA vector types (wave32) ----
typedef __attribute__((ext_vector_type(16))) __bf16 v16bf;
typedef __attribute__((ext_vector_type(8)))  __bf16 v8bf;
typedef __attribute__((ext_vector_type(2)))  __bf16 v2bf;
typedef __attribute__((ext_vector_type(8)))  float  v8f;

#define NB_   16        // batch
#define NC_   256       // channels
#define HW_   3136      // 56*56
#define W_    56
#define NG_   4         // groups
#define NTOT_ 12845056  // 16*256*3136
#define WSLICE_ 2048    // 64 cout * 32 cin per packed weight slice
#define NWELEM_ 147456  // 4 g * 9 tap * 2 cc * 2048

// split fp32 into bf16 hi + bf16 lo  (v ~= hi + lo, |lo| ~ 2^-8 |v|)
__device__ __forceinline__ void split2(float v, __bf16& h, __bf16& l) {
  h = (__bf16)v;
  l = (__bf16)(v - (float)h);
}

// ---------------------------------------------------------------------------
// One-time weight repack: w[256,64,3,3] fp32 -> packed bf16 hi/lo slices
// layout [g][tap][cc][co(64)][k(32)], so conv A-fragments are contiguous 16B.
// grid = 576, block = 256 (147456 elements).
// ---------------------------------------------------------------------------
__global__ __launch_bounds__(256) void repack_w(const float* __restrict__ w,
                                                __bf16* __restrict__ ph,
                                                __bf16* __restrict__ pl)
{
  const int i  = blockIdx.x * 256 + threadIdx.x;
  const int k  = i & 31;
  const int co = (i >> 5) & 63;
  const int cc = (i >> 11) & 1;
  const int gt = i >> 12;            // g*9 + tap
  const int tap = gt % 9;
  const int g   = gt / 9;
  const float v = w[((g * 64 + co) * 64 + cc * 32 + k) * 9 + tap];
  split2(v, ph[i], pl[i]);
}

// ---------------------------------------------------------------------------
// Grouped 3x3 conv as implicit GEMM with split-bf16 WMMA.
// grid = (784 n-tiles of 64 positions, 4 groups), block = 256 (8 waves).
// Wave (mtile = wv&3, npair = wv>>2) computes two 16(cout) x 16(pos) tiles.
// K = 64 cin * 9 taps = 576, in 18 slices of 32; 6 WMMAs per wave per slice.
// A fragments come straight from the packed (L2-resident) weights.
// ---------------------------------------------------------------------------
template<bool HAS_MASK>
__global__ __launch_bounds__(256) void conv3x3_g4_wmma(
    const float*  __restrict__ src,   // [16,256,56,56]
    const __bf16* __restrict__ pwh,   // packed hi weights
    const __bf16* __restrict__ pwl,   // packed lo weights
    const float*  __restrict__ msk,   // [16,4,7,7] (unused if !HAS_MASK)
    float* __restrict__ dst)          // [16,256,56,56]
{
  __shared__ alignas(32) __bf16 Bh[64][32];
  __shared__ alignas(32) __bf16 Bl[64][32];

  const int nt    = blockIdx.x;          // 64-position tile
  const int g     = blockIdx.y;
  const int tid   = threadIdx.x;
  const int lane  = tid & 31;
  const int wv    = tid >> 5;
  const int mtile = wv & 3;
  const int npair = wv >> 2;
  const int m     = lane & 15;
  const int hh    = lane >> 4;

  // ---- per-thread B-staging coords (invariant across taps/slices) ----
  const int n   = tid & 63;              // staged position within tile
  const int kp0 = tid >> 6;              // channel-pair base (kp = kp0 + 4j)
  {
  }
  const int pos = nt * 64 + n;
  const int b   = pos / HW_;
  const int r   = pos % HW_;
  const int y   = r / W_, x = r % W_;
  const int moff = (b * NG_ + g) * 49;
  int cb[4];
  #pragma unroll
  for (int j = 0; j < 4; ++j)
    cb[j] = (b * NC_ + g * 64 + 2 * (kp0 + 4 * j)) * HW_;

  v8f acc0 = {}, acc1 = {};
  const int coA = mtile * 16 + m;        // A-fragment row (cout within group)
  const int r0  = npair * 32 + m;        // B-fragment rows for the 2 subtiles

  for (int tap = 0; tap < 9; ++tap) {
    const int yy = y + tap / 3 - 1;
    const int xx = x + tap % 3 - 1;
    const bool inb = (unsigned)yy < (unsigned)W_ && (unsigned)xx < (unsigned)W_;
    float mval = 0.0f;
    if (HAS_MASK) { if (inb) mval = msk[moff + (yy >> 3) * 7 + (xx >> 3)]; }
    const int off = yy * W_ + xx;

    for (int cc = 0; cc < 2; ++cc) {
      __syncthreads();                   // previous slice done reading LDS
      // ---- stage B: each thread converts 4 channel pairs for position n ----
      #pragma unroll
      for (int j = 0; j < 4; ++j) {
        float v0 = 0.0f, v1 = 0.0f;
        const bool live = HAS_MASK ? (inb && mval != 0.0f) : inb;
        if (live) {
          const float* p = src + cb[j] + cc * (32 * HW_) + off;
          v0 = p[0];
          v1 = p[HW_];
          if (HAS_MASK) { v0 *= mval; v1 *= mval; }
        }
        __bf16 h0, l0, h1, l1;
        split2(v0, h0, l0);
        split2(v1, h1, l1);
        const int k = 2 * (kp0 + 4 * j);
        v2bf hp; hp.x = h0; hp.y = h1;
        v2bf lp; lp.x = l0; lp.y = l1;
        *(v2bf*)&Bh[n][k] = hp;          // single b32 LDS store
        *(v2bf*)&Bl[n][k] = lp;
      }
      __syncthreads();

      // ---- A fragments straight from packed global weights (16B loads) ----
      const int sbase = ((g * 9 + tap) * 2 + cc) * WSLICE_ + coA * 32;
      union U16 { v16bf v; v8bf p[2]; } ah, al;
      ah.p[0] = *(const v8bf*)(pwh + sbase + 16 * hh);
      ah.p[1] = *(const v8bf*)(pwh + sbase + 16 * hh + 8);
      al.p[0] = *(const v8bf*)(pwl + sbase + 16 * hh);
      al.p[1] = *(const v8bf*)(pwl + sbase + 16 * hh + 8);

      // ---- B fragments from LDS (contiguous 32B per lane) ----
      v16bf bh0 = *(const v16bf*)&Bh[r0][16 * hh];
      v16bf bl0 = *(const v16bf*)&Bl[r0][16 * hh];
      v16bf bh1 = *(const v16bf*)&Bh[r0 + 16][16 * hh];
      v16bf bl1 = *(const v16bf*)&Bl[r0 + 16][16 * hh];

      // split-bf16 product: Ah*Bh + Ah*Bl + Al*Bh (fp32 accumulate), 2 subtiles
      acc0 = __builtin_amdgcn_wmma_f32_16x16x32_bf16(false, ah.v, false, bh0, (short)0, acc0, false, false);
      acc0 = __builtin_amdgcn_wmma_f32_16x16x32_bf16(false, ah.v, false, bl0, (short)0, acc0, false, false);
      acc0 = __builtin_amdgcn_wmma_f32_16x16x32_bf16(false, al.v, false, bh0, (short)0, acc0, false, false);
      acc1 = __builtin_amdgcn_wmma_f32_16x16x32_bf16(false, ah.v, false, bh1, (short)0, acc1, false, false);
      acc1 = __builtin_amdgcn_wmma_f32_16x16x32_bf16(false, ah.v, false, bl1, (short)0, acc1, false, false);
      acc1 = __builtin_amdgcn_wmma_f32_16x16x32_bf16(false, al.v, false, bh1, (short)0, acc1, false, false);
    }
  }

  // ---- write the two 16x16 tiles (C/D: lane n, VGPR i -> M = i + 8h) ----
  #pragma unroll
  for (int t = 0; t < 2; ++t) {
    const v8f a = t ? acc1 : acc0;
    const int p2 = nt * 64 + (npair * 2 + t) * 16 + m;
    const int bb = p2 / HW_, rr = p2 % HW_;
    #pragma unroll
    for (int i = 0; i < 8; ++i) {
      const int co = g * 64 + mtile * 16 + i + 8 * hh;
      dst[(bb * NC_ + co) * HW_ + rr] = a[i];
    }
  }
}

// ---------------------------------------------------------------------------
// BN statistics, stage 1: per (channel, batch) partial sum/sumsq.
// grid = (256, 16), block = 256.
// ---------------------------------------------------------------------------
__global__ __launch_bounds__(256) void bn_partial(const float* __restrict__ src,
                                                  float* __restrict__ part)
{
  const int c = blockIdx.x, b = blockIdx.y;
  const float* p = src + (b * NC_ + c) * HW_;
  float s = 0.f, s2 = 0.f;
  for (int i = threadIdx.x; i < HW_; i += 256) {
    float v = p[i]; s += v; s2 += v * v;
  }
  __shared__ float sh[256], sq[256];
  sh[threadIdx.x] = s; sq[threadIdx.x] = s2;
  __syncthreads();
  for (int off = 128; off > 0; off >>= 1) {
    if (threadIdx.x < off) {
      sh[threadIdx.x] += sh[threadIdx.x + off];
      sq[threadIdx.x] += sq[threadIdx.x + off];
    }
    __syncthreads();
  }
  if (threadIdx.x == 0) {
    part[(c * NB_ + b) * 2 + 0] = sh[0];
    part[(c * NB_ + b) * 2 + 1] = sq[0];
  }
}

// BN statistics, stage 2: fold partials -> per-channel scale/bias. 1 block.
__global__ __launch_bounds__(256) void bn_final(const float* __restrict__ part,
                                                const float* __restrict__ gamma,
                                                const float* __restrict__ beta,
                                                float* __restrict__ sb)
{
  const int c = threadIdx.x;
  float s = 0.f, s2 = 0.f;
  #pragma unroll
  for (int b = 0; b < NB_; ++b) {
    s  += part[(c * NB_ + b) * 2 + 0];
    s2 += part[(c * NB_ + b) * 2 + 1];
  }
  const float invN = 1.0f / (float)(NB_ * HW_);
  const float mean = s * invN;
  const float var  = s2 * invN - mean * mean;
  const float sc   = gamma[c] * rsqrtf(var + 1e-5f);
  sb[2 * c]     = sc;
  sb[2 * c + 1] = beta[c] - mean * sc;
}

// buf = relu(buf*scale+bias) * mask   (in place; m2 == m1 since C unchanged)
__global__ __launch_bounds__(256) void ew_bn_relu_mask(
    float* __restrict__ buf, const float* __restrict__ sb,
    const float* __restrict__ msk)
{
  const int i = blockIdx.x * 256 + threadIdx.x;   // NTOT_ = 50176*256 exactly
  const int c = (i / HW_) % NC_;
  const int b = i / (NC_ * HW_);
  const int p = i % HW_;
  const int y = p / W_, x = p % W_;
  const int g = c >> 6;
  const float v = fmaxf(buf[i] * sb[2 * c] + sb[2 * c + 1], 0.f);
  buf[i] = v * msk[((b * NG_ + g) * 7 + (y >> 3)) * 7 + (x >> 3)];
}

// out = relu(out*scale+bias + residual)   (in place on d_out)
__global__ __launch_bounds__(256) void ew_bn_res_relu(
    float* __restrict__ out, const float* __restrict__ sb,
    const float* __restrict__ resid)
{
  const int i = blockIdx.x * 256 + threadIdx.x;
  const int c = (i / HW_) % NC_;
  const float v = out[i] * sb[2 * c] + sb[2 * c + 1] + resid[i];
  out[i] = fmaxf(v, 0.f);
}

extern "C" void kernel_launch(void* const* d_in, const int* in_sizes, int n_in,
                              void* d_out, int out_size, void* d_ws, size_t ws_size,
                              hipStream_t stream) {
  const float* x    = (const float*)d_in[0];
  const float* mask = (const float*)d_in[1];
  const float* w1   = (const float*)d_in[2];
  const float* g1   = (const float*)d_in[3];
  const float* b1   = (const float*)d_in[4];
  const float* w2   = (const float*)d_in[5];
  const float* g2   = (const float*)d_in[6];
  const float* b2   = (const float*)d_in[7];
  float* out = (float*)d_out;

  // workspace: activation buffer + BN params + BN partials + packed weights
  float*  buf   = (float*)d_ws;            // NTOT_ floats (51.4 MB)
  float*  stats = buf + (size_t)NTOT_;     // 512 floats
  float*  part  = stats + 512;             // 256*16*2 floats
  __bf16* pw1h  = (__bf16*)(part + 8192);
  __bf16* pw1l  = pw1h + NWELEM_;
  __bf16* pw2h  = pw1l + NWELEM_;
  __bf16* pw2l  = pw2h + NWELEM_;

  // one-time (per launch) weight repack to packed bf16 hi/lo
  repack_w<<<576, 256, 0, stream>>>(w1, pw1h, pw1l);
  repack_w<<<576, 256, 0, stream>>>(w2, pw2h, pw2l);

  // conv1 (mask fused, mask-zero loads skipped) -> buf
  conv3x3_g4_wmma<true><<<dim3(784, 4), 256, 0, stream>>>(x, pw1h, pw1l, mask, buf);
  // BN1 params
  bn_partial<<<dim3(256, 16), 256, 0, stream>>>(buf, part);
  bn_final<<<1, 256, 0, stream>>>(part, g1, b1, stats);
  // buf = relu(bn1(buf)) * mask  (pre-masked input of conv2)
  ew_bn_relu_mask<<<50176, 256, 0, stream>>>(buf, stats, mask);
  // conv2 -> d_out (raw)
  conv3x3_g4_wmma<false><<<dim3(784, 4), 256, 0, stream>>>(buf, pw2h, pw2l, nullptr, out);
  // BN2 params
  bn_partial<<<dim3(256, 16), 256, 0, stream>>>(out, part);
  bn_final<<<1, 256, 0, stream>>>(part, g2, b2, stats);
  // d_out = relu(bn2(d_out) + x)
  ew_bn_res_relu<<<50176, 256, 0, stream>>>(out, stats, x);
}
